// Head_266287972848
// MI455X (gfx1250) — matrix operand order repaired and checked by reading, not compiled, not run on hardware
//
#include <hip/hip_runtime.h>
#include <hip/hip_bf16.h>

typedef __attribute__((ext_vector_type(16))) _Float16 v16h;
typedef __attribute__((ext_vector_type(8)))  _Float16 v8h;
typedef __attribute__((ext_vector_type(8)))  float    v8f;

static constexpr int Bc = 8;
static constexpr int Tc = 2048;
static constexpr int Ec = 1024;
static constexpr int Hc = 64;
static constexpr float kScale = 0.125f;   // H^-0.5

__device__ __forceinline__ v16h cat16(v8h lo, v8h hi) {
    v16h r;
#pragma unroll
    for (int i = 0; i < 8; ++i) { r[i] = lo[i]; r[8 + i] = hi[i]; }
    return r;
}

// ---------------------------------------------------------------------------
// Kernel 0: W [E,H] f32  ->  WT [H,E] f16   (makes WMMA B-operand loads contiguous)
// ---------------------------------------------------------------------------
__global__ void wtrans_kernel(const float* __restrict__ W, _Float16* __restrict__ WT) {
    int id = blockIdx.x * blockDim.x + threadIdx.x;
    if (id >= Ec * Hc) return;
    int e = id / Hc;
    int h = id % Hc;
    WT[(size_t)h * Ec + e] = (_Float16)W[id];
}

// ---------------------------------------------------------------------------
// Kernel 1: projections. One wave handles 16 rows of x [B*T, E].
// Computes K,Q,V (12 WMMA tiles of 16x16 f32, K-loop over E in steps of 32).
//   kf, qf : f16 [B*T, H] row-major
//   vT     : f16 [B, H, T]   (transposed for PV B-operand)
// ---------------------------------------------------------------------------
__global__ void proj_kernel(const float* __restrict__ x,
                            const _Float16* __restrict__ WkT,
                            const _Float16* __restrict__ WqT,
                            const _Float16* __restrict__ WvT,
                            _Float16* __restrict__ kf,
                            _Float16* __restrict__ qf,
                            _Float16* __restrict__ vT) {
    const int wave = blockIdx.x * (blockDim.x >> 5) + (threadIdx.x >> 5);
    const int lane = threadIdx.x & 31;
    const int hsel = lane >> 4;     // lane half: 0 -> lanes 0-15, 1 -> lanes 16-31
    const int ln   = lane & 15;
    const int row0 = wave * 16;     // first row of this tile in [0, B*T)

    const _Float16* WT[3] = { WkT, WqT, WvT };
    v8f acc[3][4] = {};             // [weight][h-tile], f32 16x16 accumulators

    for (int e0 = 0; e0 < Ec; e0 += 32) {
        // --- A operand: x rows (f32 -> f16). Contiguous k order (this code's choice):
        //   lane ln holds row M=ln; elems 0..15 = K(e0+hsel*16+i)
        const float* xp = x + (size_t)(row0 + ln) * Ec + e0 + hsel * 16;
        v16h a;
#pragma unroll
        for (int i = 0; i < 8; ++i) a[i]     = (_Float16)xp[i];
#pragma unroll
        for (int i = 0; i < 8; ++i) a[8 + i] = (_Float16)xp[8 + i];

#pragma unroll
        for (int w = 0; w < 3; ++w) {
#pragma unroll
            for (int ht = 0; ht < 4; ++ht) {
                // B operand (32x16), same k order as A: lanes 0-15 hold K=0..15 of col N=ln,
                // lanes 16-31 hold K=16..31 -> 16 contiguous f16 from WT[h][e0+hsel*16]
                const int h = ht * 16 + ln;
                v16h bm = *(const v16h*)(WT[w] + (size_t)h * Ec + e0 + hsel * 16);
                acc[w][ht] = __builtin_amdgcn_wmma_f32_16x16x32_f16(
                    false, a, false, bm, (short)0, acc[w][ht], false, false);
            }
        }
    }

    // --- store K and Q as f16 [B*T, H] (D layout: VGPR r -> M=r(+8), lane -> N)
#pragma unroll
    for (int w = 0; w < 2; ++w) {
        _Float16* dst = (w == 0) ? kf : qf;
#pragma unroll
        for (int ht = 0; ht < 4; ++ht) {
#pragma unroll
            for (int r = 0; r < 8; ++r) {
                const int row = row0 + r + hsel * 8;
                dst[(size_t)row * Hc + ht * 16 + ln] = (_Float16)acc[w][ht][r];
            }
        }
    }

    // --- store V transposed: vT[b][h][t]; per lane the 8 r-values are 8
    // consecutive t positions -> single 16B packed store.
    const int b  = row0 / Tc;
    const int t0 = row0 % Tc;
#pragma unroll
    for (int ht = 0; ht < 4; ++ht) {
        const int h = ht * 16 + ln;
        v8h pk;
#pragma unroll
        for (int r = 0; r < 8; ++r) pk[r] = (_Float16)acc[2][ht][r];
        *(v8h*)(vT + ((size_t)b * Hc + h) * Tc + t0 + hsel * 8) = pk;
    }
}

// ---------------------------------------------------------------------------
// Kernel 2: causal flash attention. One wave per 16-query tile; online softmax
// over 32-key blocks; P staged through per-wave LDS (D-layout -> A-layout).
// ---------------------------------------------------------------------------
__global__ void attn_kernel(const _Float16* __restrict__ qf,
                            const _Float16* __restrict__ kf,
                            const _Float16* __restrict__ vT,
                            float* __restrict__ out) {
    __shared__ _Float16 smem[4][16 * 32];   // 1 KB per wave

    const int wid  = threadIdx.x >> 5;
    const int lane = threadIdx.x & 31;
    const int hsel = lane >> 4;
    const int ln   = lane & 15;

    const int tile = blockIdx.x * (blockDim.x >> 5) + wid;
    const int b    = tile / (Tc / 16);
    const int q0   = (tile % (Tc / 16)) * 16;

    // Q in A-operand layout for both 32-wide h chunks
    v16h qa[2];
#pragma unroll
    for (int c = 0; c < 2; ++c) {
        const _Float16* qp = qf + (size_t)(b * Tc + q0 + ln) * Hc + c * 32 + hsel * 16;
        qa[c] = cat16(*(const v8h*)qp, *(const v8h*)(qp + 8));
    }

    float m[8], l[8];
#pragma unroll
    for (int r = 0; r < 8; ++r) { m[r] = -3.0e38f; l[r] = 0.0f; }
    v8f acco[4] = {};

    _Float16* P = &smem[wid][0];
    const int nkb = (q0 + 47) >> 5;        // 32-key blocks covering keys <= q0+15

    for (int kb = 0; kb < nkb; ++kb) {
        const int j0 = kb * 32;
        const v8f zero = {};

        // ---- S = Q K^T : two 16-wide key tiles, K-dim = H = 64 (2 WMMAs each)
        const _Float16* kp0 = kf + (size_t)(b * Tc + j0 + ln) * Hc + hsel * 16;
        const _Float16* kp1 = kf + (size_t)(b * Tc + j0 + 16 + ln) * Hc + hsel * 16;
        v8f s0 = __builtin_amdgcn_wmma_f32_16x16x32_f16(
            false, qa[0], false, *(const v16h*)kp0, (short)0, zero, false, false);
        s0 = __builtin_amdgcn_wmma_f32_16x16x32_f16(
            false, qa[1], false, *(const v16h*)(kp0 + 32), (short)0, s0, false, false);
        v8f s1 = __builtin_amdgcn_wmma_f32_16x16x32_f16(
            false, qa[0], false, *(const v16h*)kp1, (short)0, zero, false, false);
        s1 = __builtin_amdgcn_wmma_f32_16x16x32_f16(
            false, qa[1], false, *(const v16h*)(kp1 + 32), (short)0, s1, false, false);

        // ---- scale + causal mask (D layout: VGPR r -> query row, lane -> key col)
        const bool needmask = (j0 + 31) > q0;
#pragma unroll
        for (int r = 0; r < 8; ++r) {
            float v0 = s0[r] * kScale;
            float v1 = s1[r] * kScale;
            if (needmask) {
                const int qi = q0 + r + hsel * 8;
                if (j0 + ln      > qi) v0 = -1.0e30f;
                if (j0 + 16 + ln > qi) v1 = -1.0e30f;
            }
            s0[r] = v0; s1[r] = v1;
        }

        // ---- online softmax row stats (reduce across the 16 lanes of a half)
        float corr[8];
#pragma unroll
        for (int r = 0; r < 8; ++r) {
            float mx = fmaxf(s0[r], s1[r]);
#pragma unroll
            for (int off = 8; off >= 1; off >>= 1)
                mx = fmaxf(mx, __shfl_xor(mx, off, 16));
            const float mn = fmaxf(m[r], mx);
            const float p0 = __expf(s0[r] - mn);
            const float p1 = __expf(s1[r] - mn);
            float rs = p0 + p1;
#pragma unroll
            for (int off = 8; off >= 1; off >>= 1)
                rs += __shfl_xor(rs, off, 16);
            corr[r] = __expf(m[r] - mn);
            l[r] = l[r] * corr[r] + rs;
            m[r] = mn;
            // stage P (f16) row-major [16 x 32] for A-layout readback
            const int row = r + hsel * 8;
            P[row * 32 + ln]      = (_Float16)p0;
            P[row * 32 + 16 + ln] = (_Float16)p1;
        }

        // ---- rescale O accumulators
#pragma unroll
        for (int ht = 0; ht < 4; ++ht)
#pragma unroll
            for (int r = 0; r < 8; ++r)
                acco[ht][r] *= corr[r];

        // ---- read P back in A-operand layout (same wave; compiler orders DS)
        v16h pa = cat16(*(const v8h*)(P + ln * 32 + hsel * 16),
                        *(const v8h*)(P + ln * 32 + 8 + hsel * 16));

        // ---- O += P V : B operand from vT[b][h][j0 + hsel*16 ...] (contiguous keys)
#pragma unroll
        for (int ht = 0; ht < 4; ++ht) {
            const _Float16* vp =
                vT + ((size_t)b * Hc + ht * 16 + ln) * Tc + j0 + hsel * 16;
            acco[ht] = __builtin_amdgcn_wmma_f32_16x16x32_f16(
                false, pa, false, *(const v16h*)vp, (short)0, acco[ht], false, false);
        }
    }

    // ---- epilogue: normalize and store fp32 [B, T, H]
#pragma unroll
    for (int r = 0; r < 8; ++r) {
        const float inv = 1.0f / l[r];
        const int row = b * Tc + q0 + r + hsel * 8;
#pragma unroll
        for (int ht = 0; ht < 4; ++ht)
            out[(size_t)row * Hc + ht * 16 + ln] = acco[ht][r] * inv;
    }
}

// ---------------------------------------------------------------------------
extern "C" void kernel_launch(void* const* d_in, const int* in_sizes, int n_in,
                              void* d_out, int out_size, void* d_ws, size_t ws_size,
                              hipStream_t stream) {
    const float* x  = (const float*)d_in[0];
    const float* Wk = (const float*)d_in[1];
    const float* Wq = (const float*)d_in[2];
    const float* Wv = (const float*)d_in[3];

    char* ws = (char*)d_ws;
    const size_t qkvBytes = (size_t)Bc * Tc * Hc * sizeof(_Float16);   // 2 MB each
    _Float16* qf  = (_Float16*)(ws);
    _Float16* kf  = (_Float16*)(ws + qkvBytes);
    _Float16* vT  = (_Float16*)(ws + 2 * qkvBytes);
    _Float16* WkT = (_Float16*)(ws + 3 * qkvBytes);
    _Float16* WqT = WkT + (size_t)Ec * Hc;
    _Float16* WvT = WqT + (size_t)Ec * Hc;

    // 0) weight transpose + f32->f16
    wtrans_kernel<<<(Ec * Hc + 255) / 256, 256, 0, stream>>>(Wk, WkT);
    wtrans_kernel<<<(Ec * Hc + 255) / 256, 256, 0, stream>>>(Wq, WqT);
    wtrans_kernel<<<(Ec * Hc + 255) / 256, 256, 0, stream>>>(Wv, WvT);

    // 1) projections: (B*T/16) wave-tiles, 4 waves / 128-thread block
    const int tiles = Bc * Tc / 16;            // 1024
    proj_kernel<<<tiles / 4, 128, 0, stream>>>(x, WkT, WqT, WvT, kf, qf, vT);

    // 2) flash attention: one wave per 16-query tile
    attn_kernel<<<tiles / 4, 128, 0, stream>>>(qf, kf, vT, (float*)d_out);
}
